// AttentiveFP_79164837200354
// MI455X (gfx1250) — matrix operand order repaired and checked by reading, not compiled, not run on hardware
//
#include <hip/hip_runtime.h>

typedef __attribute__((ext_vector_type(16))) _Float16 v16h;
typedef __attribute__((ext_vector_type(8)))  float    v8f;

static constexpr int N_NODES = 50000;
static constexpr int N_EDGES = 500000;
static constexpr int N_GRAPH = 2000;
static constexpr int F_IN    = 64;
static constexpr int F_H     = 128;
static constexpr int F_ED    = 16;

// ---------------- helpers ----------------
__device__ __forceinline__ unsigned ford(float f) {
  unsigned u = __float_as_uint(f);
  return (u & 0x80000000u) ? ~u : (u | 0x80000000u);
}
__device__ __forceinline__ float funord(unsigned u) {
  return __uint_as_float((u & 0x80000000u) ? (u & 0x7fffffffu) : ~u);
}
__device__ __forceinline__ float leaky01(float v) { return v >= 0.f ? v : 0.01f * v; }
__device__ __forceinline__ float sigmoidf(float v) { return 1.f / (1.f + __expf(-v)); }

#define WMMA16(a, b, c) \
  __builtin_amdgcn_wmma_f32_16x16x32_f16(false, (a), false, (b), (short)0, (c), false, false)

// Convert W[n][k] (row-major, out x in) -> B[k][n] f16, zero-padded to Kpad rows.
__global__ void k_convert_w(const float* __restrict__ W, _Float16* __restrict__ B,
                            int Kdim, int Kpad, int Ndim) {
  int i = blockIdx.x * blockDim.x + threadIdx.x;
  if (i >= Kpad * Ndim) return;
  int k = i / Ndim, n = i - k * Ndim;
  B[i] = (k < Kdim) ? (_Float16)W[(size_t)n * Kdim + k] : (_Float16)0.f;
}

// ---------------- vectorized elementwise ----------------
__global__ void k_clear_f(float* __restrict__ p, int n) {
  int i = (blockIdx.x * blockDim.x + threadIdx.x) * 4;
  if (i + 3 < n) {
    *(float4*)(p + i) = make_float4(0.f, 0.f, 0.f, 0.f);
  } else {
    for (; i < n; ++i) p[i] = 0.f;
  }
}
__global__ void k_clear_u(unsigned* __restrict__ p, int n) {
  int i = (blockIdx.x * blockDim.x + threadIdx.x) * 4;
  if (i + 3 < n) {
    *(uint4*)(p + i) = make_uint4(0u, 0u, 0u, 0u);
  } else {
    for (; i < n; ++i) p[i] = 0u;
  }
}
__global__ void k_iota(int* __restrict__ p, int n) {
  int i = blockIdx.x * blockDim.x + threadIdx.x;
  if (i < n) p[i] = i;
}
__global__ void k_elu(float* __restrict__ p, int n) {
  int i = (blockIdx.x * blockDim.x + threadIdx.x) * 4;
  if (i + 3 < n) {
    float4 v = *(const float4*)(p + i);
    v.x = v.x > 0.f ? v.x : expm1f(v.x);
    v.y = v.y > 0.f ? v.y : expm1f(v.y);
    v.z = v.z > 0.f ? v.z : expm1f(v.z);
    v.w = v.w > 0.f ? v.w : expm1f(v.w);
    *(float4*)(p + i) = v;
  } else {
    for (; i < n; ++i) { float v = p[i]; p[i] = v > 0.f ? v : expm1f(v); }
  }
}
__global__ void k_relu(float* __restrict__ p, int n) {
  int i = (blockIdx.x * blockDim.x + threadIdx.x) * 4;
  if (i + 3 < n) {
    float4 v = *(const float4*)(p + i);
    v.x = v.x > 0.f ? v.x : 0.f;
    v.y = v.y > 0.f ? v.y : 0.f;
    v.z = v.z > 0.f ? v.z : 0.f;
    v.w = v.w > 0.f ? v.w : 0.f;
    *(float4*)(p + i) = v;
  } else {
    for (; i < n; ++i) { float v = p[i]; p[i] = v > 0.f ? v : 0.f; }
  }
}

// ---------------- generic WMMA GEMM over contiguous rows ----------------
// C[M][Ncols] = act(A[M][K] @ B_f16[Kpad][Ncols] + bias); grid.x = M/16, block = (Ncols/16)*32
__global__ void k_gemm_rows(const float* __restrict__ A, int K,
                            const _Float16* __restrict__ B, int Ncols,
                            const float* __restrict__ bias,
                            float* __restrict__ C, int act) {
  const int wave = threadIdx.x >> 5, lane = threadIdx.x & 31;
  const int sel = lane >> 4, l15 = lane & 15;
  const int row0 = blockIdx.x << 4;
  const int ntile = wave << 4;
  const float* Arow = A + (size_t)(row0 + l15) * K;
  v8f c = {};
  for (int kk = 0; kk < K; kk += 32) {
    v16h a;
    const int o0 = kk + sel * 8, o1 = kk + 16 + sel * 8;
#pragma unroll
    for (int t = 0; t < 8; ++t) a[t] = (_Float16)Arow[o0 + t];
#pragma unroll
    for (int t = 0; t < 8; ++t) a[8 + t] = (_Float16)Arow[o1 + t];
    v16h b;
#pragma unroll
    for (int t = 0; t < 16; ++t) b[t] = B[(size_t)(kk + (t < 8 ? sel * 8 + t : 16 + sel * 8 + t - 8)) * Ncols + ntile + l15];
    if (kk + 32 < K)  // warm next f16 weight panel (global_prefetch_b8)
      __builtin_prefetch(B + (size_t)(kk + 32 + l15 + sel * 16) * Ncols + ntile, 0, 3);
    c = WMMA16(a, b, c);
  }
  const int col = ntile + l15;
  const float bv = bias[col];
#pragma unroll
  for (int j = 0; j < 8; ++j) {
    float v = c[j] + bv;
    if (act == 1) v = leaky01(v);
    C[(size_t)(row0 + j + sel * 8) * Ncols + col] = v;
  }
}

// ---------------- 128-wide row dots (align projections, lin2) ----------------
__global__ void k_rowdot128(const float* __restrict__ X, const float* __restrict__ w,
                            float* __restrict__ out, int M) {
  const int wave = threadIdx.x >> 5, lane = threadIdx.x & 31;
  const int row = (blockIdx.x << 3) + wave;
  if (row >= M) return;
  const float* xr = X + (size_t)row * F_H;
  float acc = 0.f;
#pragma unroll
  for (int t = 0; t < 4; ++t) acc += xr[lane + 32 * t] * w[lane + 32 * t];
#pragma unroll
  for (int off = 16; off; off >>= 1) acc += __shfl_down(acc, off, 32);
  if (lane == 0) out[row] = acc;
}

__global__ void k_lin2(const float* __restrict__ X, const float* __restrict__ w,
                       const float* __restrict__ b, float* __restrict__ out, int M) {
  const int wave = threadIdx.x >> 5, lane = threadIdx.x & 31;
  const int row = (blockIdx.x << 3) + wave;
  if (row >= M) return;
  const float* xr = X + (size_t)row * F_H;
  float acc = 0.f;
#pragma unroll
  for (int t = 0; t < 4; ++t) acc += xr[lane + 32 * t] * w[lane + 32 * t];
#pragma unroll
  for (int off = 16; off; off >>= 1) acc += __shfl_down(acc, off, 32);
  if (lane == 0) out[row] = acc + b[0];
}

// ---------------- segment softmax pieces ----------------
__global__ void k_edge_score_max(const float* __restrict__ sd, const float* __restrict__ ss,
                                 const int* __restrict__ src, const int* __restrict__ dst,
                                 const float* __restrict__ ab, float* __restrict__ score,
                                 unsigned* __restrict__ m, int E) {
  int e = blockIdx.x * blockDim.x + threadIdx.x;
  if (e >= E) return;
  float v = leaky01(sd[dst[e]] + ss[src[e]] + ab[0]);
  score[e] = v;
  atomicMax(&m[dst[e]], ford(v));
}

__global__ void k_exp_sum(float* __restrict__ score, const unsigned* __restrict__ m,
                          float* __restrict__ s, const int* __restrict__ dst, int E) {
  int e = blockIdx.x * blockDim.x + threadIdx.x;
  if (e >= E) return;
  int d = dst[e];
  float a = __expf(score[e] - funord(m[d]));
  score[e] = a;
  atomicAdd(&s[d], a);
}

// h[dst] += (e/(s[dst]+eps)) * y[src]; grid-stride, 32 lanes x 4 cols per edge,
// with global_prefetch of the next iteration's gathered row.
__global__ void k_scatter_msg(const float* __restrict__ y, const float* __restrict__ score,
                              const float* __restrict__ s, const int* __restrict__ src,
                              const int* __restrict__ dst, float* __restrict__ h, int E) {
  const int stride = gridDim.x * blockDim.x;
  const int total = E * 32;
  for (int tid = blockIdx.x * blockDim.x + threadIdx.x; tid < total; tid += stride) {
    int tn = tid + stride;
    if (tn < total)
      __builtin_prefetch(y + (size_t)src[tn >> 5] * F_H + (tn & 31) * 4, 0, 3);
    int e = tid >> 5;
    int c0 = (tid & 31) * 4;
    int d = dst[e];
    float sc = score[e] / (s[d] + 1e-16f);
    const float4 v = *(const float4*)(y + (size_t)src[e] * F_H + c0);
    float* hp = h + (size_t)d * F_H + c0;
    atomicAdd(hp + 0, v.x * sc);
    atomicAdd(hp + 1, v.y * sc);
    atomicAdd(hp + 2, v.z * sc);
    atomicAdd(hp + 3, v.w * sc);
  }
}

__global__ void k_pool(const float* __restrict__ x, const int* __restrict__ batch,
                       float* __restrict__ out, int Nn) {
  const int stride = gridDim.x * blockDim.x;
  const int total = Nn * 32;
  for (int tid = blockIdx.x * blockDim.x + threadIdx.x; tid < total; tid += stride) {
    int tn = tid + stride;
    if (tn < total)
      __builtin_prefetch(x + (size_t)(tn >> 5) * F_H + (tn & 31) * 4, 0, 3);
    int i = tid >> 5;
    int c0 = (tid & 31) * 4;
    const float4 v = *(const float4*)(x + (size_t)i * F_H + c0);
    float* op = out + (size_t)batch[i] * F_H + c0;
    atomicAdd(op + 0, v.x);
    atomicAdd(op + 1, v.y);
    atomicAdd(op + 2, v.z);
    atomicAdd(op + 3, v.w);
  }
}

// ---------------- gate conv (edge MLP) ----------------
// xj = leaky([raw[src]|edge_attr] @ nlB + nlb) into LDS f16 [16][128]; K = 80 padded to 96.
__device__ __forceinline__ void gate_compute_xj(const float* __restrict__ raw,
                                                const float* __restrict__ ea,
                                                const int* __restrict__ src, int e0,
                                                const _Float16* __restrict__ Bnl,
                                                const float* __restrict__ bnl,
                                                _Float16 (&xjs)[16][F_H]) {
  const int wave = threadIdx.x >> 5, lane = threadIdx.x & 31;
  const int sel = lane >> 4, l15 = lane & 15;
  const int e = e0 + l15;
  const float* rrow = raw + (size_t)src[e] * F_IN;
  const float* erow = ea + (size_t)e * F_ED;
  v8f c = {};
#pragma unroll
  for (int it = 0; it < 3; ++it) {
    const int kk = it * 32;
    v16h a;
    if (kk < 64) {
#pragma unroll
      for (int t = 0; t < 8; ++t) a[t] = (_Float16)rrow[kk + sel * 8 + t];
#pragma unroll
      for (int t = 0; t < 8; ++t) a[8 + t] = (_Float16)rrow[kk + 16 + sel * 8 + t];
    } else {  // k 64..79 = edge_attr, 80..95 zero-padded
#pragma unroll
      for (int t = 0; t < 8; ++t) a[t] = (_Float16)erow[sel * 8 + t];
#pragma unroll
      for (int t = 0; t < 8; ++t) a[8 + t] = (_Float16)0.f;
    }
    v16h b;
#pragma unroll
    for (int t = 0; t < 16; ++t) b[t] = Bnl[(size_t)(kk + (t < 8 ? sel * 8 + t : 16 + sel * 8 + t - 8)) * F_H + wave * 16 + l15];
    c = WMMA16(a, b, c);
  }
  const int col = wave * 16 + l15;
  const float bv = bnl[col];
#pragma unroll
  for (int j = 0; j < 8; ++j) xjs[j + sel * 8][col] = (_Float16)leaky01(c[j] + bv);
}

__global__ void k_gate_score(const float* __restrict__ raw, const float* __restrict__ ea,
                             const int* __restrict__ src, const int* __restrict__ dst,
                             const _Float16* __restrict__ Bnl, const float* __restrict__ bnl,
                             const float* __restrict__ alignw, const float* __restrict__ alignb,
                             const float* __restrict__ sd, float* __restrict__ score,
                             unsigned* __restrict__ m) {
  __shared__ _Float16 xjs[16][F_H];
  __shared__ float part[16][8];
  const int e0 = blockIdx.x << 4;
  gate_compute_xj(raw, ea, src, e0, Bnl, bnl, xjs);
  __syncthreads();
  if (threadIdx.x < 128) {
    int r = threadIdx.x >> 3, sg = threadIdx.x & 7;
    float acc = 0.f;
#pragma unroll
    for (int i = 0; i < 16; ++i) acc += (float)xjs[r][sg * 16 + i] * alignw[128 + sg * 16 + i];
    part[r][sg] = acc;
  }
  __syncthreads();
  if (threadIdx.x < 16) {
    int e = e0 + threadIdx.x;
    float acc = sd[dst[e]] + alignb[0];
#pragma unroll
    for (int i = 0; i < 8; ++i) acc += part[threadIdx.x][i];
    acc = leaky01(acc);
    score[e] = acc;
    atomicMax(&m[dst[e]], ford(acc));
  }
}

__global__ void k_gate_msg(const float* __restrict__ raw, const float* __restrict__ ea,
                           const int* __restrict__ src, const int* __restrict__ dst,
                           const _Float16* __restrict__ Bnl, const float* __restrict__ bnl,
                           const _Float16* __restrict__ Batt, const float* __restrict__ attb,
                           const float* __restrict__ score, const float* __restrict__ ssum,
                           float* __restrict__ h) {
  __shared__ _Float16 xjs[16][F_H];
  __shared__ float scale[16];
  __shared__ int dsts[16];
  const int e0 = blockIdx.x << 4;
  if (threadIdx.x < 16) {
    int e = e0 + threadIdx.x, d = dst[e];
    dsts[threadIdx.x] = d;
    scale[threadIdx.x] = score[e] / (ssum[d] + 1e-16f);
  }
  gate_compute_xj(raw, ea, src, e0, Bnl, bnl, xjs);
  __syncthreads();
  const int wave = threadIdx.x >> 5, lane = threadIdx.x & 31;
  const int sel = lane >> 4, l15 = lane & 15;
  v8f c = {};
#pragma unroll
  for (int it = 0; it < 4; ++it) {
    const int kk = it * 32;
    v16h a;
#pragma unroll
    for (int t = 0; t < 8; ++t) a[t] = xjs[l15][kk + sel * 8 + t];
#pragma unroll
    for (int t = 0; t < 8; ++t) a[8 + t] = xjs[l15][kk + 16 + sel * 8 + t];
    v16h b;
#pragma unroll
    for (int t = 0; t < 16; ++t) b[t] = Batt[(size_t)(kk + (t < 8 ? sel * 8 + t : 16 + sel * 8 + t - 8)) * F_H + wave * 16 + l15];
    c = WMMA16(a, b, c);
  }
  const int col = wave * 16 + l15;
  const float bv = attb[col];
#pragma unroll
  for (int j = 0; j < 8; ++j) {
    int r = j + sel * 8;
    atomicAdd(&h[(size_t)dsts[r] * F_H + col], (c[j] + bv) * scale[r]);
  }
}

// ---------------- fused GRU (input Xin=conv out, hidden Hprev) + ReLU ----------------
// grid.x = M/16, block = 256 (8 waves). gi/gh are 16x384 in LDS.
__global__ void k_gru(const float* __restrict__ Xin, const float* __restrict__ Hprev,
                      const _Float16* __restrict__ Bih, const float* __restrict__ bih,
                      const _Float16* __restrict__ Bhh, const float* __restrict__ bhh,
                      float* __restrict__ Xout) {
  __shared__ float gi[16][384];
  __shared__ float gh[16][384];
  const int wave = threadIdx.x >> 5, lane = threadIdx.x & 31;
  const int sel = lane >> 4, l15 = lane & 15;
  const int row0 = blockIdx.x << 4;
  for (int half = 0; half < 2; ++half) {
    const float* A = half ? Hprev : Xin;
    const _Float16* B = half ? Bhh : Bih;
    const float* bias = half ? bhh : bih;
    float(*G)[384] = half ? gh : gi;
    const float* Arow = A + (size_t)(row0 + l15) * F_H;
    for (int t = 0; t < 3; ++t) {
      const int ntile = (wave * 3 + t) * 16;  // 0..368
      v8f c = {};
#pragma unroll
      for (int it = 0; it < 4; ++it) {
        const int kk = it * 32;
        v16h a;
#pragma unroll
        for (int q = 0; q < 8; ++q) a[q] = (_Float16)Arow[kk + sel * 8 + q];
#pragma unroll
        for (int q = 0; q < 8; ++q) a[8 + q] = (_Float16)Arow[kk + 16 + sel * 8 + q];
        v16h b;
#pragma unroll
        for (int q = 0; q < 16; ++q) b[q] = B[(size_t)(kk + (q < 8 ? sel * 8 + q : 16 + sel * 8 + q - 8)) * 384 + ntile + l15];
        c = WMMA16(a, b, c);
      }
      const int col = ntile + l15;
      const float bv = bias[col];
#pragma unroll
      for (int j = 0; j < 8; ++j) G[j + sel * 8][col] = c[j] + bv;
    }
  }
  __syncthreads();
  for (int idx = threadIdx.x; idx < 16 * F_H; idx += blockDim.x) {
    int r = idx >> 7, col = idx & 127;
    float rg = sigmoidf(gi[r][col] + gh[r][col]);
    float z = sigmoidf(gi[r][128 + col] + gh[r][128 + col]);
    float n = tanhf(gi[r][256 + col] + rg * gh[r][256 + col]);
    float hp = Hprev[(size_t)(row0 + r) * F_H + col];
    float v = (1.f - z) * n + z * hp;
    Xout[(size_t)(row0 + r) * F_H + col] = v > 0.f ? v : 0.f;
  }
}

// ---------------- host orchestration ----------------
extern "C" void kernel_launch(void* const* d_in, const int* in_sizes, int n_in,
                              void* d_out, int out_size, void* d_ws, size_t ws_size,
                              hipStream_t stream) {
  (void)in_sizes; (void)n_in; (void)out_size; (void)ws_size;
  const float* raw   = (const float*)d_in[0];
  const int*   ei    = (const int*)d_in[1];
  const float* eattr = (const float*)d_in[2];
  const int*   batch = (const int*)d_in[3];
  const int* src = ei;
  const int* dst = ei + N_EDGES;

  int p = 4;
  const float* lin1_w = (const float*)d_in[p++];
  const float* lin1_b = (const float*)d_in[p++];
  const float* g_aw = (const float*)d_in[p++];   // gate align_w (1,256)
  const float* g_ab = (const float*)d_in[p++];
  const float* g_tw = (const float*)d_in[p++];   // gate att_w (128,128)
  const float* g_tb = (const float*)d_in[p++];
  const float* g_nw = (const float*)d_in[p++];   // gate nl_w (128,80)
  const float* g_nb = (const float*)d_in[p++];
  const float* gru_wih[4]; const float* gru_whh[4];
  const float* gru_bih[4]; const float* gru_bhh[4];
  gru_wih[0] = (const float*)d_in[p++]; gru_whh[0] = (const float*)d_in[p++];
  gru_bih[0] = (const float*)d_in[p++]; gru_bhh[0] = (const float*)d_in[p++];
  const float* gat_aw[2]; const float* gat_ab[2]; const float* gat_tw[2]; const float* gat_tb[2];
  for (int l = 0; l < 2; ++l) {
    gat_aw[l] = (const float*)d_in[p++]; gat_ab[l] = (const float*)d_in[p++];
    gat_tw[l] = (const float*)d_in[p++]; gat_tb[l] = (const float*)d_in[p++];
  }
  for (int l = 1; l <= 2; ++l) {
    gru_wih[l] = (const float*)d_in[p++]; gru_whh[l] = (const float*)d_in[p++];
    gru_bih[l] = (const float*)d_in[p++]; gru_bhh[l] = (const float*)d_in[p++];
  }
  const float* m_aw = (const float*)d_in[p++]; const float* m_ab = (const float*)d_in[p++];
  const float* m_tw = (const float*)d_in[p++]; const float* m_tb = (const float*)d_in[p++];
  gru_wih[3] = (const float*)d_in[p++]; gru_whh[3] = (const float*)d_in[p++];
  gru_bih[3] = (const float*)d_in[p++]; gru_bhh[3] = (const float*)d_in[p++];
  const float* lin2_w = (const float*)d_in[p++];
  const float* lin2_b = (const float*)d_in[p++];

  // workspace carve-out (256B aligned)
  char* wsp = (char*)d_ws;
  auto alloc = [&](size_t bytes) -> void* {
    void* r = (void*)wsp; wsp += (bytes + 255) & ~(size_t)255; return r;
  };
  float* xA   = (float*)alloc((size_t)N_NODES * F_H * 4);
  float* xB   = (float*)alloc((size_t)N_NODES * F_H * 4);
  float* hN   = (float*)alloc((size_t)N_NODES * F_H * 4);
  float* yN   = (float*)alloc((size_t)N_NODES * F_H * 4);
  float* score = (float*)alloc((size_t)N_EDGES * 4);
  unsigned* mN = (unsigned*)alloc((size_t)N_NODES * 4);
  float* sN  = (float*)alloc((size_t)N_NODES * 4);
  float* sdN = (float*)alloc((size_t)N_NODES * 4);
  float* ssN = (float*)alloc((size_t)N_NODES * 4);
  float* outG = (float*)alloc((size_t)N_GRAPH * F_H * 4);
  float* outB = (float*)alloc((size_t)N_GRAPH * F_H * 4);
  float* hG   = (float*)alloc((size_t)N_GRAPH * F_H * 4);
  unsigned* mG = (unsigned*)alloc((size_t)N_GRAPH * 4);
  float* sG  = (float*)alloc((size_t)N_GRAPH * 4);
  float* sdG = (float*)alloc((size_t)N_GRAPH * 4);
  int* iota  = (int*)alloc((size_t)N_NODES * 4);
  _Float16* B_lin1 = (_Float16*)alloc((size_t)64 * 128 * 2);
  _Float16* B_nl   = (_Float16*)alloc((size_t)96 * 128 * 2);
  _Float16* B_gt   = (_Float16*)alloc((size_t)128 * 128 * 2);
  _Float16* B_at[2]; B_at[0] = (_Float16*)alloc((size_t)128 * 128 * 2);
  B_at[1] = (_Float16*)alloc((size_t)128 * 128 * 2);
  _Float16* B_mt = (_Float16*)alloc((size_t)128 * 128 * 2);
  _Float16* B_ih[4]; _Float16* B_hh[4];
  for (int l = 0; l < 4; ++l) {
    B_ih[l] = (_Float16*)alloc((size_t)128 * 384 * 2);
    B_hh[l] = (_Float16*)alloc((size_t)128 * 384 * 2);
  }

  auto conv = [&](const float* W, _Float16* B, int Kd, int Kp, int Nd) {
    int n = Kp * Nd;
    k_convert_w<<<(n + 255) / 256, 256, 0, stream>>>(W, B, Kd, Kp, Nd);
  };
  auto clr_f = [&](float* q, int n) {
    k_clear_f<<<(n / 4 + 255) / 256, 256, 0, stream>>>(q, n);
  };
  auto clr_u = [&](unsigned* q, int n) {
    k_clear_u<<<(n / 4 + 255) / 256, 256, 0, stream>>>(q, n);
  };
  auto elu = [&](float* q, int n) {
    k_elu<<<(n / 4 + 255) / 256, 256, 0, stream>>>(q, n);
  };

  // weight prep
  conv(lin1_w, B_lin1, 64, 64, 128);
  conv(g_nw, B_nl, 80, 96, 128);
  conv(g_tw, B_gt, 128, 128, 128);
  conv(gat_tw[0], B_at[0], 128, 128, 128);
  conv(gat_tw[1], B_at[1], 128, 128, 128);
  conv(m_tw, B_mt, 128, 128, 128);
  for (int l = 0; l < 4; ++l) {
    conv(gru_wih[l], B_ih[l], 128, 128, 384);
    conv(gru_whh[l], B_hh[l], 128, 128, 384);
  }
  k_iota<<<(N_NODES + 255) / 256, 256, 0, stream>>>(iota, N_NODES);

  const int gN16 = N_NODES / 16, gE16 = N_EDGES / 16, gG16 = N_GRAPH / 16;
  const int gE = (N_EDGES + 255) / 256, gNe = (N_NODES + 255) / 256;
  const int SCAT_BLOCKS = 4096;  // grid-stride width for scatter/pool
  float* xcur = xA; float* xalt = xB;

  // 1) x = leaky(lin1(raw))
  k_gemm_rows<<<gN16, 256, 0, stream>>>(raw, F_IN, B_lin1, F_H, lin1_b, xcur, 1);

  // 2) gate conv
  k_rowdot128<<<N_NODES / 8, 256, 0, stream>>>(xcur, g_aw, sdN, N_NODES);
  clr_u(mN, N_NODES); clr_f(sN, N_NODES); clr_f(hN, N_NODES * F_H);
  k_gate_score<<<gE16, 256, 0, stream>>>(raw, eattr, src, dst, B_nl, g_nb, g_aw, g_ab, sdN, score, mN);
  k_exp_sum<<<gE, 256, 0, stream>>>(score, mN, sN, dst, N_EDGES);
  k_gate_msg<<<gE16, 256, 0, stream>>>(raw, eattr, src, dst, B_nl, g_nb, B_gt, g_tb, score, sN, hN);
  elu(hN, N_NODES * F_H);
  k_gru<<<gN16, 256, 0, stream>>>(hN, xcur, B_ih[0], gru_bih[0], B_hh[0], gru_bhh[0], xalt);
  { float* t = xcur; xcur = xalt; xalt = t; }

  // 3) two GAT + GRU layers
  for (int l = 0; l < 2; ++l) {
    k_rowdot128<<<N_NODES / 8, 256, 0, stream>>>(xcur, gat_aw[l], sdN, N_NODES);
    k_rowdot128<<<N_NODES / 8, 256, 0, stream>>>(xcur, gat_aw[l] + 128, ssN, N_NODES);
    k_gemm_rows<<<gN16, 256, 0, stream>>>(xcur, F_H, B_at[l], F_H, gat_tb[l], yN, 0);
    clr_u(mN, N_NODES); clr_f(sN, N_NODES); clr_f(hN, N_NODES * F_H);
    k_edge_score_max<<<gE, 256, 0, stream>>>(sdN, ssN, src, dst, gat_ab[l], score, mN, N_EDGES);
    k_exp_sum<<<gE, 256, 0, stream>>>(score, mN, sN, dst, N_EDGES);
    k_scatter_msg<<<SCAT_BLOCKS, 256, 0, stream>>>(yN, score, sN, src, dst, hN, N_EDGES);
    elu(hN, N_NODES * F_H);
    k_gru<<<gN16, 256, 0, stream>>>(hN, xcur, B_ih[1 + l], gru_bih[1 + l], B_hh[1 + l],
                                    gru_bhh[1 + l], xalt);
    { float* t = xcur; xcur = xalt; xalt = t; }
  }

  // 4) global add pool + relu
  clr_f(outG, N_GRAPH * F_H);
  k_pool<<<SCAT_BLOCKS, 256, 0, stream>>>(xcur, batch, outG, N_NODES);
  k_relu<<<(N_GRAPH * F_H / 4 + 255) / 256, 256, 0, stream>>>(outG, N_GRAPH * F_H);

  // 5) molecular GAT + GRU timesteps (edges: node i -> graph batch[i])
  k_rowdot128<<<N_NODES / 8, 256, 0, stream>>>(xcur, m_aw + 128, ssN, N_NODES);
  k_gemm_rows<<<gN16, 256, 0, stream>>>(xcur, F_H, B_mt, F_H, m_tb, yN, 0);
  float* ocur = outG; float* oalt = outB;
  for (int t = 0; t < 2; ++t) {
    k_rowdot128<<<N_GRAPH / 8, 256, 0, stream>>>(ocur, m_aw, sdG, N_GRAPH);
    clr_u(mG, N_GRAPH); clr_f(sG, N_GRAPH); clr_f(hG, N_GRAPH * F_H);
    k_edge_score_max<<<gNe, 256, 0, stream>>>(sdG, ssN, iota, batch, m_ab, score, mG, N_NODES);
    k_exp_sum<<<gNe, 256, 0, stream>>>(score, mG, sG, batch, N_NODES);
    k_scatter_msg<<<SCAT_BLOCKS, 256, 0, stream>>>(yN, score, sG, iota, batch, hG, N_NODES);
    elu(hG, N_GRAPH * F_H);
    k_gru<<<gG16, 256, 0, stream>>>(hG, ocur, B_ih[3], gru_bih[3], B_hh[3], gru_bhh[3], oalt);
    { float* tt = ocur; ocur = oalt; oalt = tt; }
  }

  // 6) final linear
  k_lin2<<<N_GRAPH / 8, 256, 0, stream>>>(ocur, lin2_w, lin2_b, (float*)d_out, N_GRAPH);
}
